// conv_block_odconv_84456236908885
// MI455X (gfx1250) — matrix operand order repaired and checked by reading, not compiled, not run on hardware
//
#include <hip/hip_runtime.h>
#include <hip/hip_bf16.h>

typedef _Float16 v8h  __attribute__((ext_vector_type(8)));
typedef _Float16 v16h __attribute__((ext_vector_type(16)));
typedef float    v8f  __attribute__((ext_vector_type(8)));

#define BN_EPS 1e-5f

__device__ __forceinline__ float sigmoidf_(float x) { return 1.f / (1.f + expf(-x)); }

__device__ __forceinline__ v16h mk16(const _Float16* lo, const _Float16* hi) {
  v8h a = *(const v8h*)lo;
  v8h b = *(const v8h*)hi;
  return __builtin_shufflevector(a, b, 0, 1, 2, 3, 4, 5, 6, 7, 8, 9, 10, 11, 12, 13, 14, 15);
}

// ---------------- global average pool over one [H*W] plane per block ----------------
__global__ __launch_bounds__(256) void pool_kernel(const float* __restrict__ x,
                                                   float* __restrict__ pooled) {
  __shared__ float red[256];
  size_t base = (size_t)blockIdx.x * 16384;
  float s = 0.f;
  for (int i = threadIdx.x; i < 16384; i += 256) s += x[base + i];
  red[threadIdx.x] = s; __syncthreads();
  for (int o = 128; o > 0; o >>= 1) {
    if (threadIdx.x < o) red[threadIdx.x] += red[threadIdx.x + o];
    __syncthreads();
  }
  if (threadIdx.x == 0) pooled[blockIdx.x] = red[0] * (1.f / 16384.f);
}

// ---------------- attention head: fc -> batchBN -> relu -> 4 heads ----------------
__global__ __launch_bounds__(256) void attn_kernel(
    const float* __restrict__ pooled,                 // [16][C]
    const float* __restrict__ fc_w,                   // [16][C]
    const float* __restrict__ bn_g, const float* __restrict__ bn_b,   // [16]
    const float* __restrict__ ch_w, const float* __restrict__ ch_b,   // [Cch][16],[Cch]
    const float* __restrict__ fl_w, const float* __restrict__ fl_b,   // [128][16],[128]
    const float* __restrict__ sp_w, const float* __restrict__ sp_b,   // [9][16],[9]
    const float* __restrict__ kn_w, const float* __restrict__ kn_b,   // [4][16],[4]
    float* __restrict__ ch_o, float* __restrict__ fl_o,
    float* __restrict__ sp_o, float* __restrict__ kn_o,
    int C, int Cch) {
  __shared__ float hsh[16][16];
  __shared__ float mArr[16], vArr[16];
  __shared__ float kl[16][4];
  const int tid = threadIdx.x;
  const int b = tid >> 4, a = tid & 15;

  float s = 0.f;
  for (int ci = 0; ci < C; ++ci) s += pooled[b * C + ci] * fc_w[a * C + ci];
  hsh[b][a] = s;
  __syncthreads();

  if (tid < 16) {
    float m = 0.f;
    for (int i = 0; i < 16; ++i) m += hsh[i][tid];
    m *= (1.f / 16.f);
    float v = 0.f;
    for (int i = 0; i < 16; ++i) { float d = hsh[i][tid] - m; v += d * d; }
    v *= (1.f / 16.f);
    mArr[tid] = m; vArr[tid] = v;
  }
  __syncthreads();

  float hv = (hsh[b][a] - mArr[a]) * rsqrtf(vArr[a] + BN_EPS) * bn_g[a] + bn_b[a];
  hv = fmaxf(hv, 0.f);
  hsh[b][a] = hv;   // each thread owns its own cell
  __syncthreads();

  for (int idx = tid; idx < 16 * Cch; idx += 256) {
    int bb = idx / Cch, c = idx % Cch;
    float t = ch_b[c];
    for (int j = 0; j < 16; ++j) t += hsh[bb][j] * ch_w[c * 16 + j];
    ch_o[idx] = sigmoidf_(t);
  }
  for (int idx = tid; idx < 16 * 128; idx += 256) {
    int bb = idx >> 7, c = idx & 127;
    float t = fl_b[c];
    for (int j = 0; j < 16; ++j) t += hsh[bb][j] * fl_w[c * 16 + j];
    fl_o[idx] = sigmoidf_(t);
  }
  if (tid < 144) {
    int bb = tid / 9, j9 = tid % 9;
    float t = sp_b[j9];
    for (int j = 0; j < 16; ++j) t += hsh[bb][j] * sp_w[j9 * 16 + j];
    sp_o[tid] = sigmoidf_(t);
  }
  if (tid < 64) {
    int bb = tid >> 2, k = tid & 3;
    float t = kn_b[k];
    for (int j = 0; j < 16; ++j) t += hsh[bb][j] * kn_w[k * 16 + j];
    kl[bb][k] = t;
  }
  __syncthreads();
  if (tid < 16) {
    float mx = fmaxf(fmaxf(kl[tid][0], kl[tid][1]), fmaxf(kl[tid][2], kl[tid][3]));
    float e0 = expf(kl[tid][0] - mx), e1 = expf(kl[tid][1] - mx);
    float e2 = expf(kl[tid][2] - mx), e3 = expf(kl[tid][3] - mx);
    float inv = 1.f / (e0 + e1 + e2 + e3);
    kn_o[tid * 4 + 0] = e0 * inv; kn_o[tid * 4 + 1] = e1 * inv;
    kn_o[tid * 4 + 2] = e2 * inv; kn_o[tid * 4 + 3] = e3 * inv;
  }
}

// ---------------- aggregate per-sample kernels -> f16 [b][co][tap][ci] ----------------
__global__ __launch_bounds__(256) void aggw_kernel(const float* __restrict__ W, // [4][128][CIN][3][3]
                                                   const float* __restrict__ sp, // [16][9]
                                                   const float* __restrict__ kn, // [16][4]
                                                   _Float16* __restrict__ agg, int CIN) {
  int b = blockIdx.x >> 7, co = blockIdx.x & 127;
  float k0 = kn[b * 4 + 0], k1 = kn[b * 4 + 1], k2 = kn[b * 4 + 2], k3 = kn[b * 4 + 3];
  size_t stride = (size_t)128 * CIN * 9;
  for (int e = threadIdx.x; e < 9 * CIN; e += 256) {
    int tap = e / CIN, ci = e % CIN;
    size_t wi = ((size_t)co * CIN + ci) * 9 + tap;
    float s = (k0 * W[wi] + k1 * W[wi + stride] + k2 * W[wi + 2 * stride] + k3 * W[wi + 3 * stride])
              * sp[b * 9 + tap];
    agg[((size_t)(b * 128 + co) * 9 + tap) * CIN + ci] = (_Float16)s;
  }
}

// ---------------- ODConv core: shifted implicit GEMM via v_wmma_f32_16x16x32_f16 -------
// One workgroup per (output row gy, sample b). 8 waves; wave w owns
// Cout rows [32*(w>>1), 32*(w>>1)+32) x pixels [64*(w&1), 64*(w&1)+64).
// Two A fragments share every B fragment -> 1 ds_load_b128 per WMMA.
// LDS tile: [3 rows][132 cols][72 ch] f16 (57 KB), zero-padded halo, ch_att fused.
template <int CIN, typename T>
__global__ __launch_bounds__(256) void odconv_kernel(
    const T* __restrict__ xin,                 // [16][CIN][128][128]
    const _Float16* __restrict__ aggw,         // [16][128][9][CIN] f16
    const float* __restrict__ ch_att,          // [16][CIN]
    const float* __restrict__ fl_att,          // [16][128]
    float* __restrict__ out) {                 // [16][128][128][128]
  constexpr int CPAD = 72;                     // col stride = 36 dwords: conflict-free, 16B aligned
  __shared__ __align__(16) _Float16 lds[3 * 132 * CPAD];
  __shared__ float chs[64];

  const int gy = blockIdx.x, b = blockIdx.y, tid = threadIdx.x;
  const int wave = tid >> 5, lane = tid & 31;
  const int laneM = lane & 15, laneHi = lane >> 4;
  const int co0 = (wave >> 1) * 32;
  const int px0 = (wave & 1) * 64;

  const v8f vzero = {0.f, 0.f, 0.f, 0.f, 0.f, 0.f, 0.f, 0.f};
  v8f acc0[4], acc1[4];
#pragma unroll
  for (int i = 0; i < 4; ++i) { acc0[i] = vzero; acc1[i] = vzero; }

  const _Float16* Abase0 = aggw + (size_t)(b * 128 + co0 + laneM) * 9 * CIN;
  const _Float16* Abase1 = Abase0 + (size_t)16 * 9 * CIN;

#pragma unroll 1
  for (int p = 0; p < CIN / 64; ++p) {
    // ---- channel attention for this 64-channel slab ----
    if (tid < 64) chs[tid] = ch_att[b * CIN + p * 64 + tid];
    __syncthreads();
    // ---- stage 64 ch x 3 halo rows x 130 cols, f16, 8 channels per ds_store_b128 ----
    for (int e = tid; e < 8 * 390; e += 256) {
      int cgrp = e / 390;
      int rem = e % 390;
      int lrow = rem / 130;
      int lcol = rem % 130;
      int gx = lcol - 1, gyy = gy + lrow - 1;
      const bool inb = (gx >= 0 && gx < 128 && gyy >= 0 && gyy < 128);
      const int cib = p * 64 + cgrp * 8;
      v8h pk;
#pragma unroll
      for (int j = 0; j < 8; ++j) {
        float v = 0.f;
        if (inb)
          v = (float)xin[((size_t)(b * CIN + cib + j) * 128 + gyy) * 128 + gx] * chs[cgrp * 8 + j];
        pk[j] = (_Float16)v;
      }
      *(v8h*)&lds[(lrow * 132 + lcol) * CPAD + cgrp * 8] = pk;
    }
    __syncthreads();

    // ---- 9 taps x 2 K-chunks of 32 channels ----
#pragma unroll 1
    for (int tap = 0; tap < 9; ++tap) {
      const int dy = tap / 3, dx = tap % 3;
#pragma unroll
      for (int cc = 0; cc < 2; ++cc) {
        const int ci0l = cc * 32;
        // A fragments: lanes 0-15 K 0..15, lanes 16-31 K 16..31
        const _Float16* Ap0 = Abase0 + tap * CIN + p * 64 + ci0l + laneHi * 16;
        const _Float16* Ap1 = Abase1 + tap * CIN + p * 64 + ci0l + laneHi * 16;
        v16h afrag0 = mk16(Ap0, Ap0 + 8);
        v16h afrag1 = mk16(Ap1, Ap1 + 8);
        // B base: lanes 0-15 K 0..15, lanes 16-31 K 16..31, 16 contiguous channels
        const int boff = (dy * 132 + dx + laneM + px0) * CPAD + ci0l + laneHi * 16;
        v16h bcur = mk16(&lds[boff], &lds[boff] + 8);
#pragma unroll
        for (int n0 = 0; n0 < 4; ++n0) {
          v16h bnext = bcur;
          if (n0 < 3) {
            const _Float16* Bp = &lds[boff + (n0 + 1) * 16 * CPAD];
            bnext = mk16(Bp, Bp + 8);
          }
          acc0[n0] = __builtin_amdgcn_wmma_f32_16x16x32_f16(
              false, afrag0, false, bcur, (short)0, acc0[n0], false, false);
          acc1[n0] = __builtin_amdgcn_wmma_f32_16x16x32_f16(
              false, afrag1, false, bcur, (short)0, acc1[n0], false, false);
          bcur = bnext;
        }
      }
    }
    __syncthreads();
  }

  // ---- store, filter attention fused. VGPR r: M=r (lanes 0-15) / M=8+r (lanes 16-31) ----
#pragma unroll
  for (int n0 = 0; n0 < 4; ++n0) {
    const int px = px0 + n0 * 16 + laneM;
#pragma unroll
    for (int r = 0; r < 8; ++r) {
      const int coA = co0 + laneHi * 8 + r;
      const int coB = coA + 16;
      out[((size_t)(b * 128 + coA) * 128 + gy) * 128 + px] = acc0[n0][r] * fl_att[b * 128 + coA];
      out[((size_t)(b * 128 + coB) * 128 + gy) * 128 + px] = acc1[n0][r] * fl_att[b * 128 + coB];
    }
  }
}

// ---------------- batch-norm statistics: one channel per block ----------------
__global__ __launch_bounds__(256) void bnstats_kernel(const float* __restrict__ y,
                                                      float* __restrict__ mv) {
  __shared__ float r1[256], r2[256];
  const int c = blockIdx.x, tid = threadIdx.x;
  float s = 0.f, q = 0.f;
  for (int b = 0; b < 16; ++b) {
    const float* p = y + ((size_t)b * 128 + c) * 16384;
    for (int i = tid; i < 16384; i += 256) { float v = p[i]; s += v; q += v * v; }
  }
  r1[tid] = s; r2[tid] = q; __syncthreads();
  for (int o = 128; o > 0; o >>= 1) {
    if (tid < o) { r1[tid] += r1[tid + o]; r2[tid] += r2[tid + o]; }
    __syncthreads();
  }
  if (tid == 0) {
    const float n = 16.f * 16384.f;
    float m = r1[0] / n;
    mv[c] = m;
    mv[128 + c] = r2[0] / n - m * m;
  }
}

// ---------------- BN+ReLU; optional f32 out (in-place ok), f16 out, fused pool --------
__global__ __launch_bounds__(256) void bnrelu_kernel(const float* __restrict__ y,
                                                     const float* __restrict__ mv,
                                                     const float* __restrict__ g,
                                                     const float* __restrict__ bb,
                                                     float* __restrict__ outf,
                                                     _Float16* __restrict__ outh,
                                                     float* __restrict__ pooled) {
  __shared__ float red[256];
  const int bc = blockIdx.x, tid = threadIdx.x;
  const int c = bc & 127;
  const float m = mv[c], v = mv[128 + c];
  const float sc = g[c] * rsqrtf(v + BN_EPS), sh = bb[c] - m * sc;
  const float* p = y + (size_t)bc * 16384;
  float s = 0.f;
  for (int i = tid; i < 16384; i += 256) {
    float val = fmaxf(p[i] * sc + sh, 0.f);
    if (outf) outf[(size_t)bc * 16384 + i] = val;
    if (outh) outh[(size_t)bc * 16384 + i] = (_Float16)val;
    s += val;
  }
  if (pooled) {
    red[tid] = s; __syncthreads();
    for (int o = 128; o > 0; o >>= 1) {
      if (tid < o) red[tid] += red[tid + o];
      __syncthreads();
    }
    if (tid == 0) pooled[bc] = red[0] * (1.f / 16384.f);
  }
}

// ---------------- workspace layout (bytes) ----------------
static constexpr size_t OFF_AGG1 = 0;                         // 16*128*9*64*2   = 2,359,296
static constexpr size_t OFF_AGG2 = 2359296;                   // 16*128*9*128*2  = 4,718,592
static constexpr size_t OFF_H1   = 7077888;                   // 16*128*128*128*2= 67,108,864
static constexpr size_t OFF_P1   = 74186752;                  // 16*64*4
static constexpr size_t OFF_P2   = 74190848;                  // 16*128*4
static constexpr size_t OFF_CH1  = 74199040;                  // 16*64*4
static constexpr size_t OFF_FL1  = 74203136;                  // 16*128*4
static constexpr size_t OFF_SP1  = 74211328;                  // 16*9*4
static constexpr size_t OFF_KN1  = 74212352;                  // 16*4*4
static constexpr size_t OFF_CH2  = 74212864;                  // 16*128*4
static constexpr size_t OFF_FL2  = 74221056;                  // 16*128*4
static constexpr size_t OFF_SP2  = 74229248;
static constexpr size_t OFF_KN2  = 74230272;
static constexpr size_t OFF_ST1  = 74230784;                  // 2*128*4
static constexpr size_t OFF_ST2  = 74231808;                  // 2*128*4

extern "C" void kernel_launch(void* const* d_in, const int* in_sizes, int n_in,
                              void* d_out, int out_size, void* d_ws, size_t ws_size,
                              hipStream_t stream) {
  (void)in_sizes; (void)n_in; (void)out_size; (void)ws_size;
  const float* x      = (const float*)d_in[0];
  const float* w1     = (const float*)d_in[1];
  const float* a_fc   = (const float*)d_in[2];
  const float* a_bn_g = (const float*)d_in[3];
  const float* a_bn_b = (const float*)d_in[4];
  const float* a_ch_w = (const float*)d_in[5];
  const float* a_ch_b = (const float*)d_in[6];
  const float* a_fl_w = (const float*)d_in[7];
  const float* a_fl_b = (const float*)d_in[8];
  const float* a_sp_w = (const float*)d_in[9];
  const float* a_sp_b = (const float*)d_in[10];
  const float* a_kn_w = (const float*)d_in[11];
  const float* a_kn_b = (const float*)d_in[12];
  const float* bn1_g  = (const float*)d_in[13];
  const float* bn1_b  = (const float*)d_in[14];
  const float* w2     = (const float*)d_in[15];
  const float* b_fc   = (const float*)d_in[16];
  const float* b_bn_g = (const float*)d_in[17];
  const float* b_bn_b = (const float*)d_in[18];
  const float* b_ch_w = (const float*)d_in[19];
  const float* b_ch_b = (const float*)d_in[20];
  const float* b_fl_w = (const float*)d_in[21];
  const float* b_fl_b = (const float*)d_in[22];
  const float* b_sp_w = (const float*)d_in[23];
  const float* b_sp_b = (const float*)d_in[24];
  const float* b_kn_w = (const float*)d_in[25];
  const float* b_kn_b = (const float*)d_in[26];
  const float* bn2_g  = (const float*)d_in[27];
  const float* bn2_b  = (const float*)d_in[28];

  char* ws = (char*)d_ws;
  _Float16* agg1 = (_Float16*)(ws + OFF_AGG1);
  _Float16* agg2 = (_Float16*)(ws + OFF_AGG2);
  _Float16* h1   = (_Float16*)(ws + OFF_H1);
  float* pooled1 = (float*)(ws + OFF_P1);
  float* pooled2 = (float*)(ws + OFF_P2);
  float* ch1 = (float*)(ws + OFF_CH1); float* fl1 = (float*)(ws + OFF_FL1);
  float* sp1 = (float*)(ws + OFF_SP1); float* kn1 = (float*)(ws + OFF_KN1);
  float* ch2 = (float*)(ws + OFF_CH2); float* fl2 = (float*)(ws + OFF_FL2);
  float* sp2 = (float*)(ws + OFF_SP2); float* kn2 = (float*)(ws + OFF_KN2);
  float* st1 = (float*)(ws + OFF_ST1); float* st2 = (float*)(ws + OFF_ST2);
  float* y = (float*)d_out;

  // ---- layer 1 ----
  pool_kernel<<<dim3(16 * 64), 256, 0, stream>>>(x, pooled1);
  attn_kernel<<<1, 256, 0, stream>>>(pooled1, a_fc, a_bn_g, a_bn_b,
                                     a_ch_w, a_ch_b, a_fl_w, a_fl_b,
                                     a_sp_w, a_sp_b, a_kn_w, a_kn_b,
                                     ch1, fl1, sp1, kn1, 64, 64);
  aggw_kernel<<<dim3(16 * 128), 256, 0, stream>>>(w1, sp1, kn1, agg1, 64);
  odconv_kernel<64, float><<<dim3(128, 16), 256, 0, stream>>>(x, agg1, ch1, fl1, y);
  bnstats_kernel<<<128, 256, 0, stream>>>(y, st1);
  bnrelu_kernel<<<2048, 256, 0, stream>>>(y, st1, bn1_g, bn1_b, nullptr, h1, pooled2);

  // ---- layer 2 ----
  attn_kernel<<<1, 256, 0, stream>>>(pooled2, b_fc, b_bn_g, b_bn_b,
                                     b_ch_w, b_ch_b, b_fl_w, b_fl_b,
                                     b_sp_w, b_sp_b, b_kn_w, b_kn_b,
                                     ch2, fl2, sp2, kn2, 128, 128);
  aggw_kernel<<<dim3(16 * 128), 256, 0, stream>>>(w2, sp2, kn2, agg2, 128);
  odconv_kernel<128, _Float16><<<dim3(128, 16), 256, 0, stream>>>(h1, agg2, ch2, fl2, y);
  bnstats_kernel<<<128, 256, 0, stream>>>(y, st2);
  bnrelu_kernel<<<2048, 256, 0, stream>>>(y, st2, bn2_g, bn2_b, y, nullptr, nullptr);
}
